// PointGPTBackbone_41979010351628
// MI455X (gfx1250) — matrix operand order repaired and checked by reading, not compiled, not run on hardware
//
#include <hip/hip_runtime.h>
#include <math.h>

typedef __bf16 bf16;
typedef __attribute__((ext_vector_type(16))) __bf16 v16bf;
typedef __attribute__((ext_vector_type(8)))  __bf16 v8bf;
typedef __attribute__((ext_vector_type(8)))  float  v8f;

#define B_   16
#define N_   8192
#define G_   128
#define K_   32
#define C_   384
#define H_   6
#define L_   12
#define HID_ 1536
#define DH_  64
#define EPS_ 1e-5f
#define ROWS_ (B_*G_*K_)   /* 65536 encoder rows */
#define TOK_  (B_*G_)      /* 2048 tokens */

// ---------------------------------------------------------------- utilities
__global__ void conv_bf16_kernel(const float* __restrict__ in, bf16* __restrict__ out, long long n) {
  long long i = (long long)blockIdx.x * blockDim.x + threadIdx.x;
  if (i < n) out[i] = (bf16)in[i];
}

__global__ void addpos_kernel(const float* __restrict__ h, const float* __restrict__ pos,
                              float* __restrict__ x, long long n) {
  long long i = (long long)blockIdx.x * blockDim.x + threadIdx.x;
  if (i < n) x[i] = h[i] + pos[i];
}

// ---------------------------------------------------------------- FPS (grid=B, block=256)
__global__ __launch_bounds__(256) void fps_kernel(const float* __restrict__ xyz, int* __restrict__ cent_idx) {
  int b = blockIdx.x, tid = threadIdx.x;
  const float* base = xyz + (size_t)b * N_ * 3;
  float dist[N_ / 256];
#pragma unroll
  for (int j = 0; j < N_ / 256; ++j) dist[j] = 1e10f;
  __shared__ float sc[3];
  __shared__ float sv[256];
  __shared__ int   si[256];
  __shared__ int   s_cent;
  int cent = 0;
  for (int it = 0; it < G_; ++it) {
    if (tid == 0) cent_idx[b * G_ + it] = cent;
    if (tid < 3) sc[tid] = base[cent * 3 + tid];
    __syncthreads();
    float cx = sc[0], cy = sc[1], cz = sc[2];
    float bv = -1.f; int bi = 0;
    for (int j = 0; j < N_ / 256; ++j) {
      int idx = tid + j * 256;
      float dx = base[idx * 3] - cx, dy = base[idx * 3 + 1] - cy, dz = base[idx * 3 + 2] - cz;
      float d = dx * dx + dy * dy + dz * dz;
      float nd = fminf(dist[j], d);
      dist[j] = nd;
      if (nd > bv || (nd == bv && idx < bi)) { bv = nd; bi = idx; }
    }
    sv[tid] = bv; si[tid] = bi;
    __syncthreads();
    for (int s = 128; s > 0; s >>= 1) {
      if (tid < s) {
        if (sv[tid + s] > sv[tid] || (sv[tid + s] == sv[tid] && si[tid + s] < si[tid])) {
          sv[tid] = sv[tid + s]; si[tid] = si[tid + s];
        }
      }
      __syncthreads();
    }
    if (tid == 0) s_cent = si[0];
    __syncthreads();
    cent = s_cent;
  }
}

// -------------------------------------------------- kNN + gather (grid=B*G, block=256)
__global__ __launch_bounds__(256) void knn_kernel(const float* __restrict__ xyz, const int* __restrict__ cent_idx,
                                                  float* __restrict__ center_raw, float* __restrict__ neigh_raw) {
  int bg = blockIdx.x; int b = bg / G_; int tid = threadIdx.x;
  const float* base = xyz + (size_t)b * N_ * 3;
  __shared__ float sc[3];
  __shared__ float sv[256];
  __shared__ int   si[256];
  __shared__ int   s_list[K_];
  if (tid == 0) {
    int ci = cent_idx[bg];
    sc[0] = base[ci * 3]; sc[1] = base[ci * 3 + 1]; sc[2] = base[ci * 3 + 2];
  }
  __syncthreads();
  float cx = sc[0], cy = sc[1], cz = sc[2];
  float d[N_ / 256];
#pragma unroll
  for (int j = 0; j < N_ / 256; ++j) {
    int idx = tid + j * 256;
    float dx = base[idx * 3] - cx, dy = base[idx * 3 + 1] - cy, dz = base[idx * 3 + 2] - cz;
    d[j] = dx * dx + dy * dy + dz * dz;
  }
  for (int kk = 0; kk < K_; ++kk) {
    float bv = 3.4e38f; int bi = 1 << 30;
    for (int j = 0; j < N_ / 256; ++j) {
      int idx = tid + j * 256;
      if (d[j] < bv || (d[j] == bv && idx < bi)) { bv = d[j]; bi = idx; }
    }
    sv[tid] = bv; si[tid] = bi;
    __syncthreads();
    for (int s = 128; s > 0; s >>= 1) {
      if (tid < s) {
        if (sv[tid + s] < sv[tid] || (sv[tid + s] == sv[tid] && si[tid + s] < si[tid])) {
          sv[tid] = sv[tid + s]; si[tid] = si[tid + s];
        }
      }
      __syncthreads();
    }
    if (tid == 0) s_list[kk] = si[0];
    __syncthreads();
    int win = s_list[kk];
    if ((win & 255) == tid) d[win >> 8] = 3.4e38f;
    __syncthreads();
  }
  if (tid < K_) {
    int idx = s_list[tid];
    for (int c = 0; c < 3; ++c)
      neigh_raw[((size_t)bg * K_ + tid) * 3 + c] = base[idx * 3 + c] - sc[c];
  }
  if (tid < 3) center_raw[bg * 3 + tid] = sc[tid];
}

// ------------------------------------------- greedy nearest-path order (grid=B, block=128)
__global__ __launch_bounds__(128) void order_kernel(const float* __restrict__ center_raw, int* __restrict__ order) {
  int b = blockIdx.x, tid = threadIdx.x;
  __shared__ float cxs[G_], cys[G_], czs[G_];
  __shared__ int vis[G_];
  __shared__ float sv[G_];
  __shared__ int si[G_];
  __shared__ int s_last;
  cxs[tid] = center_raw[(b * G_ + tid) * 3];
  cys[tid] = center_raw[(b * G_ + tid) * 3 + 1];
  czs[tid] = center_raw[(b * G_ + tid) * 3 + 2];
  vis[tid] = (tid == 0) ? 1 : 0;
  if (tid == 0) { order[b * G_] = 0; s_last = 0; }
  __syncthreads();
  for (int step = 1; step < G_; ++step) {
    int last = s_last;
    float dd;
    if (vis[tid]) dd = 3.4e38f;
    else {
      float dx = cxs[tid] - cxs[last], dy = cys[tid] - cys[last], dz = czs[tid] - czs[last];
      dd = dx * dx + dy * dy + dz * dz;
    }
    sv[tid] = dd; si[tid] = tid;
    __syncthreads();
    for (int s = 64; s > 0; s >>= 1) {
      if (tid < s) {
        if (sv[tid + s] < sv[tid] || (sv[tid + s] == sv[tid] && si[tid + s] < si[tid])) {
          sv[tid] = sv[tid + s]; si[tid] = si[tid + s];
        }
      }
      __syncthreads();
    }
    if (tid == 0) { int w = si[0]; order[b * G_ + step] = w; s_last = w; vis[w] = 1; }
    __syncthreads();
  }
}

// ------------------------------------------------ apply ordering (grid=B*G, block=96)
__global__ void reorder_kernel(const float* __restrict__ center_raw, const float* __restrict__ neigh_raw,
                               const int* __restrict__ order, float* __restrict__ center_o,
                               float* __restrict__ neigh_o) {
  int bg = blockIdx.x, tid = threadIdx.x;
  int b = bg / G_, gp = bg % G_;
  int src = order[b * G_ + gp];
  int k = tid / 3, c = tid % 3;
  neigh_o[((size_t)bg * K_ + k) * 3 + c] = neigh_raw[(((size_t)b * G_ + src) * K_ + k) * 3 + c];
  if (tid < 3) center_o[bg * 3 + tid] = center_raw[(b * G_ + src) * 3 + tid];
}

// ------------------------------------------------ encoder stage 1 (K=3 pointwise)
__global__ void enc1_kernel(const float* __restrict__ neigh_o, const float* __restrict__ w,
                            const float* __restrict__ bv, float* __restrict__ f1) {
  int row = blockIdx.x, c = threadIdx.x;  // block 128
  const float* x = neigh_o + (size_t)row * 3;
  f1[(size_t)row * 128 + c] = x[0] * w[c * 3] + x[1] * w[c * 3 + 1] + x[2] * w[c * 3 + 2] + bv[c];
}

// ------------------------------------------------ BatchNorm stats (grid=C, block=256)
__global__ __launch_bounds__(256) void bnstats_kernel(const float* __restrict__ f, int rows, int Cch,
                                                      float* __restrict__ mean, float* __restrict__ rstd) {
  int ch = blockIdx.x, tid = threadIdx.x;
  float s = 0.f, ss = 0.f;
  for (int r = tid; r < rows; r += 256) {
    float v = f[(size_t)r * Cch + ch];
    s += v; ss += v * v;
  }
  __shared__ float s1[256], s2[256];
  s1[tid] = s; s2[tid] = ss;
  __syncthreads();
  for (int st = 128; st > 0; st >>= 1) {
    if (tid < st) { s1[tid] += s1[tid + st]; s2[tid] += s2[tid + st]; }
    __syncthreads();
  }
  if (tid == 0) {
    float m = s1[0] / rows;
    float var = s2[0] / rows - m * m;
    mean[ch] = m;
    rstd[ch] = rsqrtf(var + EPS_);
  }
}

__global__ void bnapply_kernel(const float* __restrict__ f, const float* __restrict__ mean,
                               const float* __restrict__ rstd, const float* __restrict__ g,
                               const float* __restrict__ bb, bf16* __restrict__ out,
                               long long n, int Cch) {
  long long i = (long long)blockIdx.x * blockDim.x + threadIdx.x;
  if (i >= n) return;
  int ch = (int)(i % Cch);
  float v = (f[i] - mean[ch]) * rstd[ch] * g[ch] + bb[ch];
  out[i] = (bf16)fmaxf(v, 0.f);
}

// ------------------------------------------------ WMMA GEMM:  C = act(alpha*A.Wt + bias) [+resid]
// A: (M,K) bf16 row-major (lda), W: (N,K) bf16 row-major (ldw).
// Register-blocked: each wave computes a 32x64 macro-tile (2 M-tiles x 4 N-tiles, 8 f32x8 accs).
// Per K-step: 4 A-frag loads + 8 B-frag loads feed 8 WMMAs (1.5 b128/WMMA); A-frags reused 4x.
// Requires M%32==0, N%64==0, K%32==0 (true for every GEMM in this model).
__global__ __launch_bounds__(128) void gemm_bf16_kernel(
    const bf16* __restrict__ A, long long sA, int lda,
    const bf16* __restrict__ W, long long sW, int ldw,
    float* __restrict__ C, int ldc, long long cOuter, int cInner, long long cInnerStr,
    bf16* __restrict__ C16,
    const float* __restrict__ bias, const float* __restrict__ resid,
    float alpha, int act, int M, int Nn, int Kk) {
  int wave = threadIdx.x >> 5;
  int lane = threadIdx.x & 31;
  int ntn = Nn >> 6;             // 64-wide N macro-tiles
  int nt  = (M >> 5) * ntn;      // 32-tall M macro-tiles
  int tile = blockIdx.x * 4 + wave;
  if (tile >= nt) return;
  int tm = tile / ntn;
  int tn = tile - tm * ntn;
  long long y = blockIdx.y;
  const bf16* Ab = A + y * sA;
  const bf16* Wb = W + y * sW;
  long long cOff = (y / cInner) * cOuter + (y % cInner) * cInnerStr;
  int kh = lane >> 4;       // half-select
  int mn = lane & 15;       // row (A) / column (B)
  const bf16* arow0 = Ab + (long long)(tm * 32 + mn) * lda;        // M-tile 0
  const bf16* arow1 = arow0 + (long long)16 * lda;                  // M-tile 1
  const bf16* wrow  = Wb + (long long)(tn * 64 + mn) * ldw + kh * 8;
  long long wstep = (long long)16 * ldw;                            // next N-tile
  v8f acc[8];
#pragma unroll
  for (int i = 0; i < 8; ++i) acc[i] = {};
  for (int k0 = 0; k0 < Kk; k0 += 32) {
    // A fragments (ISA layout): lane<16 holds K = k0+[0..7] & k0+16+[0..7]; lane>=16 shifts by 8
    v8bf a0lo = *(const v8bf*)(arow0 + k0 + 8 * kh);
    v8bf a0hi = *(const v8bf*)(arow0 + k0 + 16 + 8 * kh);
    v8bf a1lo = *(const v8bf*)(arow1 + k0 + 8 * kh);
    v8bf a1hi = *(const v8bf*)(arow1 + k0 + 16 + 8 * kh);
    v16bf a0, a1;
#pragma unroll
    for (int i = 0; i < 8; ++i) {
      a0[i] = a0lo[i]; a0[8 + i] = a0hi[i];
      a1[i] = a1lo[i]; a1[8 + i] = a1hi[i];
    }
#pragma unroll
    for (int n = 0; n < 4; ++n) {
      // B fragment: lane holds column n16 = lane&15, K = k0+8*(lane>>4)+[0..7] & k0+16+8*(lane>>4)+[0..7]
      v8bf blo = *(const v8bf*)(wrow + (long long)n * wstep + k0);
      v8bf bhi = *(const v8bf*)(wrow + (long long)n * wstep + k0 + 16);
      v16bf bm;
#pragma unroll
      for (int i = 0; i < 8; ++i) { bm[i] = blo[i]; bm[8 + i] = bhi[i]; }
      acc[n]     = __builtin_amdgcn_wmma_f32_16x16x32_bf16(false, a0, false, bm, (short)0, acc[n],     false, false);
      acc[4 + n] = __builtin_amdgcn_wmma_f32_16x16x32_bf16(false, a1, false, bm, (short)0, acc[4 + n], false, false);
    }
  }
#pragma unroll
  for (int mi = 0; mi < 2; ++mi) {
#pragma unroll
    for (int n = 0; n < 4; ++n) {
      int col = tn * 64 + n * 16 + mn;
      float bs = bias ? bias[col] : 0.f;
      v8f av = acc[mi * 4 + n];
#pragma unroll
      for (int r = 0; r < 8; ++r) {
        int row = tm * 32 + mi * 16 + kh * 8 + r;
        float v = av[r] * alpha + bs;
        if (act == 1) v = 0.5f * v * (1.f + erff(v * 0.70710678118654752f));  // exact GELU
        long long idx = cOff + (long long)row * ldc + col;
        if (resid) v += resid[idx];
        C[idx] = v;
        if (C16) C16[idx] = (bf16)v;
      }
    }
  }
}

// --------------------------------- max-pool + concat (grid=B*G, block=256)
__global__ __launch_bounds__(256) void maxcat_kernel(const float* __restrict__ f2, bf16* __restrict__ cat) {
  int bg = blockIdx.x, ch = threadIdx.x;
  const float* fr = f2 + (size_t)bg * K_ * 256;
  float m = -3.4e38f;
  for (int k = 0; k < K_; ++k) m = fmaxf(m, fr[k * 256 + ch]);
  bf16 mb = (bf16)m;
  for (int k = 0; k < K_; ++k) {
    size_t base = ((size_t)bg * K_ + k) * 512;
    cat[base + ch] = mb;
    cat[base + 256 + ch] = (bf16)fr[k * 256 + ch];
  }
}

// --------------------------------- token max-pool (grid=B*G, block=384)
__global__ void tokenmax_kernel(const float* __restrict__ f4, float* __restrict__ tokens) {
  int bg = blockIdx.x, ch = threadIdx.x;
  const float* fr = f4 + (size_t)bg * K_ * C_;
  float m = -3.4e38f;
  for (int k = 0; k < K_; ++k) m = fmaxf(m, fr[k * C_ + ch]);
  tokens[(size_t)bg * C_ + ch] = m;
}

// --------------------------------- sine positional embedding (grid=B*G, block=192)
__global__ void pos_kernel(const float* __restrict__ center_o, float* __restrict__ pos) {
  int bg = blockIdx.x, tid = threadIdx.x;
  int i = tid >> 6, j = tid & 63;
  float inv = powf(10000.f, -((float)j) / 64.f);
  float ang = center_o[bg * 3 + i] * inv;
  pos[(size_t)bg * C_ + i * 128 + 2 * j]     = sinf(ang);
  pos[(size_t)bg * C_ + i * 128 + 2 * j + 1] = cosf(ang);
}

// --------------------------------- sos-shifted token seed (grid=B*G, block=384)
__global__ void seed_kernel(const float* __restrict__ tokens, const float* __restrict__ sos,
                            float* __restrict__ h) {
  int bg = blockIdx.x, c = threadIdx.x;
  int g = bg % G_;
  h[(size_t)bg * C_ + c] = (g == 0) ? sos[c] : tokens[((size_t)bg - 1) * C_ + c];
}

// --------------------------------- LayerNorm over C=384 (grid=rows, block=128)
__global__ __launch_bounds__(128) void ln_kernel(const float* __restrict__ X, const float* __restrict__ g,
                                                 const float* __restrict__ b, float* __restrict__ outF,
                                                 bf16* __restrict__ outB) {
  int row = blockIdx.x, tid = threadIdx.x;
  const float* x = X + (size_t)row * C_;
  float loc[3];
  float s = 0.f;
#pragma unroll
  for (int i = 0; i < 3; ++i) { loc[i] = x[tid + i * 128]; s += loc[i]; }
  __shared__ float red[128];
  red[tid] = s; __syncthreads();
  for (int st = 64; st > 0; st >>= 1) { if (tid < st) red[tid] += red[tid + st]; __syncthreads(); }
  float mean = red[0] / C_;
  __syncthreads();
  float ss = 0.f;
#pragma unroll
  for (int i = 0; i < 3; ++i) { float dv = loc[i] - mean; ss += dv * dv; }
  red[tid] = ss; __syncthreads();
  for (int st = 64; st > 0; st >>= 1) { if (tid < st) red[tid] += red[tid + st]; __syncthreads(); }
  float rstd = rsqrtf(red[0] / C_ + EPS_);
#pragma unroll
  for (int i = 0; i < 3; ++i) {
    int c = tid + i * 128;
    float v = (loc[i] - mean) * rstd * g[c] + b[c];
    if (outF) outF[(size_t)row * C_ + c] = v;
    if (outB) outB[(size_t)row * C_ + c] = (bf16)v;
  }
}

// --------------------------------- unpack qkv -> q,k bf16 (b,h,s,d) and vT bf16 (b,h,d,s)
__global__ void unpack_kernel(const float* __restrict__ qkv, bf16* __restrict__ qb,
                              bf16* __restrict__ kb, bf16* __restrict__ vT) {
  int row = blockIdx.x, tid = threadIdx.x;  // block 384
  int b = row / G_, s = row % G_;
  int h = tid / DH_, d = tid % DH_;
  const float* q3 = qkv + (size_t)row * (3 * C_);
  size_t qi = (((size_t)b * H_ + h) * G_ + s) * DH_ + d;
  qb[qi] = (bf16)q3[tid];
  kb[qi] = (bf16)q3[C_ + tid];
  vT[(((size_t)b * H_ + h) * DH_ + d) * G_ + s] = (bf16)q3[2 * C_ + tid];
}

// --------------------------------- causal softmax (grid=B*H*G, block=128)
__global__ __launch_bounds__(128) void softmax_kernel(const float* __restrict__ scores, bf16* __restrict__ attn) {
  int idx = blockIdx.x;
  int q = idx % G_;
  int tid = threadIdx.x;
  float v = scores[(size_t)idx * G_ + tid];
  if (tid > q) v = -1e9f;
  __shared__ float red[128];
  red[tid] = v; __syncthreads();
  for (int st = 64; st > 0; st >>= 1) { if (tid < st) red[tid] = fmaxf(red[tid], red[tid + st]); __syncthreads(); }
  float m = red[0];
  __syncthreads();
  float e = expf(v - m);
  red[tid] = e; __syncthreads();
  for (int st = 64; st > 0; st >>= 1) { if (tid < st) red[tid] += red[tid + st]; __syncthreads(); }
  attn[(size_t)idx * G_ + tid] = (bf16)(e / red[0]);
}

// ================================================================ host orchestration
extern "C" void kernel_launch(void* const* d_in, const int* in_sizes, int n_in,
                              void* d_out, int out_size, void* d_ws, size_t ws_size,
                              hipStream_t stream) {
  (void)in_sizes; (void)n_in; (void)out_size; (void)ws_size;
  const float* xyz    = (const float*)d_in[0];
  const float* ec1_w  = (const float*)d_in[1];
  const float* ec1_b  = (const float*)d_in[2];
  const float* bn1_g  = (const float*)d_in[3];
  const float* bn1_b  = (const float*)d_in[4];
  const float* ec2_w  = (const float*)d_in[5];
  const float* ec2_b  = (const float*)d_in[6];
  const float* sc1_w  = (const float*)d_in[7];
  const float* sc1_b  = (const float*)d_in[8];
  const float* bn2_g  = (const float*)d_in[9];
  const float* bn2_b  = (const float*)d_in[10];
  const float* sc2_w  = (const float*)d_in[11];
  const float* sc2_b  = (const float*)d_in[12];
  const float* sos    = (const float*)d_in[13];
  const float* ln1_g  = (const float*)d_in[14];
  const float* ln1_b  = (const float*)d_in[15];
  const float* qkv_w  = (const float*)d_in[16];
  const float* qkv_b  = (const float*)d_in[17];
  const float* out_w  = (const float*)d_in[18];
  const float* out_b  = (const float*)d_in[19];
  const float* ln2_g  = (const float*)d_in[20];
  const float* ln2_b  = (const float*)d_in[21];
  const float* mlp1_w = (const float*)d_in[22];
  const float* mlp1_b = (const float*)d_in[23];
  const float* mlp2_w = (const float*)d_in[24];
  const float* mlp2_b = (const float*)d_in[25];
  const float* lnf_g  = (const float*)d_in[26];
  const float* lnf_b  = (const float*)d_in[27];

  char* wsb = (char*)d_ws;
  size_t off = 0;
  auto alloc = [&](size_t bytes) -> char* {
    off = (off + 255) & ~(size_t)255;
    char* p = wsb + off;
    off += bytes;
    return p;
  };

  int*   cent_idx   = (int*)  alloc(B_ * G_ * 4);
  float* center_raw = (float*)alloc(B_ * G_ * 3 * 4);
  float* neigh_raw  = (float*)alloc((size_t)B_ * G_ * K_ * 3 * 4);
  int*   order      = (int*)  alloc(B_ * G_ * 4);
  float* center_o   = (float*)alloc(B_ * G_ * 3 * 4);
  float* neigh_o    = (float*)alloc((size_t)B_ * G_ * K_ * 3 * 4);
  float* mean       = (float*)alloc(512 * 4);
  float* rstd       = (float*)alloc(512 * 4);
  float* tokens     = (float*)alloc((size_t)TOK_ * C_ * 4);
  float* pos        = (float*)alloc((size_t)TOK_ * C_ * 4);
  float* hbuf       = (float*)alloc((size_t)TOK_ * C_ * 4);
  float* xbuf       = (float*)alloc((size_t)TOK_ * C_ * 4);
  bf16*  y_bf       = (bf16*) alloc((size_t)TOK_ * C_ * 2);
  float* qkvf       = (float*)alloc((size_t)TOK_ * 3 * C_ * 4);
  bf16*  q_bf       = (bf16*) alloc((size_t)TOK_ * C_ * 2);
  bf16*  k_bf       = (bf16*) alloc((size_t)TOK_ * C_ * 2);
  bf16*  vT_bf      = (bf16*) alloc((size_t)TOK_ * C_ * 2);
  float* scores     = (float*)alloc((size_t)B_ * H_ * G_ * G_ * 4);
  bf16*  attn_bf    = (bf16*) alloc((size_t)B_ * H_ * G_ * G_ * 2);
  float* obuf       = (float*)alloc((size_t)TOK_ * C_ * 4);
  bf16*  o_bf       = (bf16*) alloc((size_t)TOK_ * C_ * 2);
  float* hmid       = (float*)alloc((size_t)TOK_ * HID_ * 4);
  bf16*  hmid_bf    = (bf16*) alloc((size_t)TOK_ * HID_ * 2);
  // bf16 weights
  bf16* ec2w_bf = (bf16*)alloc(256 * 128 * 2);
  bf16* sc1w_bf = (bf16*)alloc(512 * 512 * 2);
  bf16* sc2w_bf = (bf16*)alloc(384 * 512 * 2);
  bf16* qkvw_bf = (bf16*)alloc((size_t)L_ * 3 * C_ * C_ * 2);
  bf16* outw_bf = (bf16*)alloc((size_t)L_ * C_ * C_ * 2);
  bf16* m1w_bf  = (bf16*)alloc((size_t)L_ * HID_ * C_ * 2);
  bf16* m2w_bf  = (bf16*)alloc((size_t)L_ * C_ * HID_ * 2);
  // big reusable encoder regions (lifetime-disjoint aliases)
  char* bufA = alloc((size_t)ROWS_ * 512 * 4);  // f1 (f32 x128) then pre3 (f32 x512)
  char* bufB = alloc((size_t)ROWS_ * 512 * 2);  // a1_bf (x128) then cat_bf (x512) then a3_bf (x512)
  char* bufC = alloc((size_t)ROWS_ * 384 * 4);  // f2 (f32 x256) then f4 (f32 x384)
  float* f1     = (float*)bufA;
  float* pre3   = (float*)bufA;
  bf16*  a1_bf  = (bf16*)bufB;
  bf16*  cat_bf = (bf16*)bufB;
  bf16*  a3_bf  = (bf16*)bufB;
  float* f2     = (float*)bufC;
  float* f4     = (float*)bufC;

  auto conv = [&](const float* in, bf16* out, long long n) {
    conv_bf16_kernel<<<dim3((unsigned)((n + 255) / 256)), 256, 0, stream>>>(in, out, n);
  };
  auto gemm = [&](const bf16* A, long long sA, int lda, const bf16* W, long long sW, int ldw,
                  float* Cp, int ldc, long long cOuter, int cInner, long long cInnerStr,
                  bf16* C16, const float* bias, const float* resid, float alpha, int act,
                  int M, int Nn, int Kk, int batch) {
    int nt = (M / 32) * (Nn / 64);  // 32x64 macro-tiles, 4 waves/block
    dim3 grid((unsigned)((nt + 3) / 4), (unsigned)batch);
    gemm_bf16_kernel<<<grid, 128, 0, stream>>>(A, sA, lda, W, sW, ldw, Cp, ldc, cOuter, cInner,
                                               cInnerStr, C16, bias, resid, alpha, act, M, Nn, Kk);
  };

  // ---- weight conversion (once per call; deterministic) ----
  conv(ec2_w, ec2w_bf, 256 * 128);
  conv(sc1_w, sc1w_bf, 512 * 512);
  conv(sc2_w, sc2w_bf, 384 * 512);
  conv(qkv_w, qkvw_bf, (long long)L_ * 3 * C_ * C_);
  conv(out_w, outw_bf, (long long)L_ * C_ * C_);
  conv(mlp1_w, m1w_bf, (long long)L_ * HID_ * C_);
  conv(mlp2_w, m2w_bf, (long long)L_ * C_ * HID_);

  // ---- grouping ----
  fps_kernel<<<B_, 256, 0, stream>>>(xyz, cent_idx);
  knn_kernel<<<B_ * G_, 256, 0, stream>>>(xyz, cent_idx, center_raw, neigh_raw);
  order_kernel<<<B_, 128, 0, stream>>>(center_raw, order);
  reorder_kernel<<<B_ * G_, 96, 0, stream>>>(center_raw, neigh_raw, order, center_o, neigh_o);

  // ---- encoder ----
  enc1_kernel<<<ROWS_, 128, 0, stream>>>(neigh_o, ec1_w, ec1_b, f1);
  bnstats_kernel<<<128, 256, 0, stream>>>(f1, ROWS_, 128, mean, rstd);
  {
    long long n = (long long)ROWS_ * 128;
    bnapply_kernel<<<dim3((unsigned)((n + 255) / 256)), 256, 0, stream>>>(f1, mean, rstd, bn1_g, bn1_b, a1_bf, n, 128);
  }
  gemm(a1_bf, 0, 128, ec2w_bf, 0, 128, f2, 256, 0, 1, 0, nullptr, ec2_b, nullptr, 1.f, 0, ROWS_, 256, 128, 1);
  maxcat_kernel<<<TOK_, 256, 0, stream>>>(f2, cat_bf);
  gemm(cat_bf, 0, 512, sc1w_bf, 0, 512, pre3, 512, 0, 1, 0, nullptr, sc1_b, nullptr, 1.f, 0, ROWS_, 512, 512, 1);
  bnstats_kernel<<<512, 256, 0, stream>>>(pre3, ROWS_, 512, mean, rstd);
  {
    long long n = (long long)ROWS_ * 512;
    bnapply_kernel<<<dim3((unsigned)((n + 255) / 256)), 256, 0, stream>>>(pre3, mean, rstd, bn2_g, bn2_b, a3_bf, n, 512);
  }
  gemm(a3_bf, 0, 512, sc2w_bf, 0, 512, f4, C_, 0, 1, 0, nullptr, sc2_b, nullptr, 1.f, 0, ROWS_, C_, 512, 1);
  tokenmax_kernel<<<TOK_, C_, 0, stream>>>(f4, tokens);

  // ---- GPT extractor ----
  pos_kernel<<<TOK_, 192, 0, stream>>>(center_o, pos);
  seed_kernel<<<TOK_, C_, 0, stream>>>(tokens, sos, hbuf);

  const long long nTC = (long long)TOK_ * C_;
  float* hp = hbuf;
  float* xp = xbuf;
  for (int l = 0; l < L_; ++l) {
    addpos_kernel<<<dim3((unsigned)((nTC + 255) / 256)), 256, 0, stream>>>(hp, pos, xp, nTC);
    ln_kernel<<<TOK_, 128, 0, stream>>>(xp, ln1_g + l * C_, ln1_b + l * C_, nullptr, y_bf);
    gemm(y_bf, 0, C_, qkvw_bf + (size_t)l * 3 * C_ * C_, 0, C_, qkvf, 3 * C_, 0, 1, 0,
         nullptr, qkv_b + l * 3 * C_, nullptr, 1.f, 0, TOK_, 3 * C_, C_, 1);
    unpack_kernel<<<TOK_, C_, 0, stream>>>(qkvf, q_bf, k_bf, vT_bf);
    // scores[bh] = q @ k^T / sqrt(dh)
    gemm(q_bf, (long long)G_ * DH_, DH_, k_bf, (long long)G_ * DH_, DH_, scores, G_,
         (long long)G_ * G_, 1, 0, nullptr, nullptr, nullptr, 0.125f, 0, G_, G_, DH_, B_ * H_);
    softmax_kernel<<<B_ * H_ * G_, 128, 0, stream>>>(scores, attn_bf);
    // o[b, q, h*64+d] = attn @ v   (W = v^T, row-major (64,128))
    gemm(attn_bf, (long long)G_ * G_, G_, vT_bf, (long long)DH_ * G_, G_, obuf, C_,
         (long long)G_ * C_, H_, DH_, o_bf, nullptr, nullptr, 1.f, 0, G_, DH_, G_, B_ * H_);
    // x = x + o @ ow^T + ob
    gemm(o_bf, 0, C_, outw_bf + (size_t)l * C_ * C_, 0, C_, xp, C_, 0, 1, 0,
         nullptr, out_b + l * C_, xp, 1.f, 0, TOK_, C_, C_, 1);
    ln_kernel<<<TOK_, 128, 0, stream>>>(xp, ln2_g + l * C_, ln2_b + l * C_, nullptr, y_bf);
    gemm(y_bf, 0, C_, m1w_bf + (size_t)l * HID_ * C_, 0, C_, hmid, HID_, 0, 1, 0,
         hmid_bf, mlp1_b + l * HID_, nullptr, 1.f, 1 /*gelu*/, TOK_, HID_, C_, 1);
    gemm(hmid_bf, 0, HID_, m2w_bf + (size_t)l * C_ * HID_, 0, HID_, xp, C_, 0, 1, 0,
         nullptr, mlp2_b + l * C_, xp, 1.f, 0, TOK_, C_, HID_, 1);
    float* t = hp; hp = xp; xp = t;  // new h = xp result
  }
  // final LayerNorm -> d_out (f32, B*G*C)
  ln_kernel<<<TOK_, 128, 0, stream>>>(hp, lnf_g, lnf_b, (float*)d_out, nullptr);
}
